// HGT_39307540693517
// MI455X (gfx1250) — matrix operand order repaired and checked by reading, not compiled, not run on hardware
//
#include <hip/hip_runtime.h>
#include <cmath>

// ---------------- problem constants (from reference) ----------------
#define HID    64
#define HEADS  4
#define HDX    16
#define NLAY   5
#define NI     65536
#define NV     32768
#define NBASE  16384
#define EFLOW  262144
#define EUSES  262144
#define EDEF   262144
#define EBASE  131072
#define NGR    64
#define NPER   1024
#define POOLK  16

// Input index map. Assumption: harness flattens setup_inputs() dict in
// insertion order at top level, with the nested `params` pytree flattened
// JAX-style (dict keys alphabetical, lists in order).
enum {
  IN_X_INSTR = 0, IN_X_VAR = 1, IN_X_BASE = 2,
  IN_A_REL_DEF = 3, IN_A_REL_FLOW = 4, IN_A_REL_USES = 5, IN_A_REL_BASE = 6,
  IN_GNN_REL_B = 7, IN_GNN_REL_W = 8, IN_GNN_ROOT_W = 9,
  IN_JK_IN_B = 10, IN_JK_IN_W = 11, IN_JK_OUT_B = 12, IN_JK_OUT_W = 13,
  IN_KQV_B_INSTR = 14, IN_KQV_B_VAR = 15, IN_KQV_B_BASE = 16,
  IN_KQV_W_INSTR = 17, IN_KQV_W_VAR = 18, IN_KQV_W_BASE = 19,
  IN_M_REL_DEF = 20, IN_M_REL_FLOW = 21, IN_M_REL_USES = 22, IN_M_REL_BASE = 23,
  IN_MLP_B0 = 24, /* ..33 */
  IN_MLP_W0 = 34, /* ..43 */
  IN_OUT_B_INSTR = 44, IN_OUT_B_VAR = 45,
  IN_OUT_W_INSTR = 46, IN_OUT_W_VAR = 47,
  IN_P_REL_DEF = 48, IN_P_REL_FLOW = 49, IN_P_REL_USES = 50, IN_P_REL_BASE = 51,
  IN_SKIP_INSTR = 52, IN_SKIP_VAR = 53,
  IN_EI_FLOW = 54, IN_EI_USES = 55, IN_EI_DEF = 56, IN_EI_BASE = 57, IN_BATCH = 58
};

typedef __attribute__((ext_vector_type(16))) _Float16 v16h;
typedef __attribute__((ext_vector_type(8)))  float    v8f;

__device__ __forceinline__ float hgt_gelu(float x) {
  return 0.5f * x * (1.0f + erff(x * 0.7071067811865476f));
}
// order-preserving float<->int for atomicMax-based segment max
__device__ __forceinline__ int   hgt_enc(float f) { int i = __float_as_int(f); return (i >= 0) ? i : (i ^ 0x7fffffff); }
__device__ __forceinline__ float hgt_dec(int i)   { return __int_as_float((i >= 0) ? i : (i ^ 0x7fffffff)); }

// ---------------- WMMA GEMM: C[M,N] = A[M,K] @ W[K,N] + bias ----------------
// One 16x16 output tile per wave32; f16 in / f32 accumulate via
// v_wmma_f32_16x16x32_f16. GUARD=false fast path: M%16==0, N%16==0, K%32==0 ->
// unconditional float4 (b128) A loads, lane-coalesced B loads, no exec churn.
// Activations (GELU) are hoisted to a separate elementwise pass so the GEMM
// inner loop stays pure load -> cvt_pk_f16 -> wmma.
template <bool GUARD>
__global__ void hgt_gemm_wmma(const float* __restrict__ A, int lda,
                              const float* __restrict__ W, int ldw,
                              const float* __restrict__ bias,
                              float* __restrict__ C, int ldc,
                              int M, int K, int N)
{
  const int lane = threadIdx.x & 31;
  const int wpb  = blockDim.x >> 5;
  const int wave = blockIdx.x * wpb + (threadIdx.x >> 5);
  const int nMt = (M + 15) >> 4, nNt = (N + 15) >> 4;
  const int nT = nMt * nNt;
  const int totW = gridDim.x * wpb;
  const int half = lane >> 4, l15 = lane & 15;

  for (int t = wave; t < nT; t += totW) {
    const int mt = t / nNt, nt = t - mt * nNt;
    const int m0 = mt << 4, n0 = nt << 4;
    const int row = m0 + l15;            // A row held by this lane
    const int col = n0 + l15;            // B/C column held by this lane
    v8f acc = {};
    if (!GUARD) {
      const float* arow = A + (size_t)row * lda;
      const float* wcol = W + col;
      for (int k0 = 0; k0 < K; k0 += 32) {
        if (k0 + 32 < K) __builtin_prefetch(arow + k0 + 32, 0, 3);
        const float4 a0 = *(const float4*)(arow + k0 + half * 8);
        const float4 a1 = *(const float4*)(arow + k0 + half * 8 + 4);
        const float4 a2 = *(const float4*)(arow + k0 + 16 + half * 8);
        const float4 a3 = *(const float4*)(arow + k0 + 16 + half * 8 + 4);
        v16h av, bv;
        av[0]=(_Float16)a0.x; av[1]=(_Float16)a0.y; av[2]=(_Float16)a0.z; av[3]=(_Float16)a0.w;
        av[4]=(_Float16)a1.x; av[5]=(_Float16)a1.y; av[6]=(_Float16)a1.z; av[7]=(_Float16)a1.w;
        av[8]=(_Float16)a2.x; av[9]=(_Float16)a2.y; av[10]=(_Float16)a2.z; av[11]=(_Float16)a2.w;
        av[12]=(_Float16)a3.x; av[13]=(_Float16)a3.y; av[14]=(_Float16)a3.z; av[15]=(_Float16)a3.w;
        #pragma unroll
        for (int j = 0; j < 16; ++j)     // j<8 -> rows k0+half*8+j; j>=8 -> rows k0+16+half*8+(j-8)
          bv[j] = (_Float16)wcol[(size_t)(k0 + (j < 8 ? half * 8 + j : 16 + half * 8 + (j - 8))) * ldw];
        acc = __builtin_amdgcn_wmma_f32_16x16x32_f16(false, av, false, bv, (short)0, acc, false, false);
      }
      const float bcol = bias ? bias[col] : 0.0f;
      #pragma unroll
      for (int r = 0; r < 8; ++r)
        C[(size_t)(m0 + half * 8 + r) * ldc + col] = acc[r] + bcol;
    } else {
      const bool rowOK = row < M, colOK = col < N;
      for (int k0 = 0; k0 < K; k0 += 32) {
        v16h av, bv;
        #pragma unroll
        for (int i = 0; i < 8; ++i) {
          const int ka = k0 + half * 8 + i;
          const int kb = k0 + 16 + half * 8 + i;
          av[i]     = (_Float16)((rowOK && ka < K) ? A[(size_t)row * lda + ka] : 0.0f);
          av[i + 8] = (_Float16)((rowOK && kb < K) ? A[(size_t)row * lda + kb] : 0.0f);
        }
        #pragma unroll
        for (int j = 0; j < 16; ++j) {
          const int kk = k0 + (j < 8 ? half * 8 + j : 16 + half * 8 + (j - 8));
          bv[j] = (_Float16)((colOK && kk < K) ? W[(size_t)kk * ldw + col] : 0.0f);
        }
        acc = __builtin_amdgcn_wmma_f32_16x16x32_f16(false, av, false, bv, (short)0, acc, false, false);
      }
      const float bcol = (bias && colOK) ? bias[col] : 0.0f;
      #pragma unroll
      for (int r = 0; r < 8; ++r) {
        const int orow = m0 + half * 8 + r;
        if (orow < M && colOK) C[(size_t)orow * ldc + col] = acc[r] + bcol;
      }
    }
  }
}

// ---------------- misc small kernels ----------------
__global__ void hgt_fill_f32(float* p, float v, int n) {
  int t = blockIdx.x * blockDim.x + threadIdx.x; if (t < n) p[t] = v;
}
__global__ void hgt_fill_i32(int* p, int v, int n) {
  int t = blockIdx.x * blockDim.x + threadIdx.x; if (t < n) p[t] = v;
}

// in-place exact GELU, vectorized (n must be a multiple of 4 — always true here)
__global__ void hgt_gelu_inplace(float* __restrict__ p, int n4) {
  int t = blockIdx.x * blockDim.x + threadIdx.x;
  if (t >= n4) return;
  float4 v = ((const float4*)p)[t];
  v.x = hgt_gelu(v.x); v.y = hgt_gelu(v.y); v.z = hgt_gelu(v.z); v.w = hgt_gelu(v.w);
  ((float4*)p)[t] = v;
}

// build 64x64 block-diagonal weight from per-head 16x16 blocks [4][16][16]
__global__ void hgt_blockdiag(const float* __restrict__ A, float* __restrict__ w) {
  int t = blockIdx.x * blockDim.x + threadIdx.x;
  if (t >= 64 * 64) return;
  const int r = t >> 6, c = t & 63;
  const int hr = r >> 4, hc = c >> 4;
  w[t] = (hr == hc) ? A[hr * 256 + (r & 15) * 16 + (c & 15)] : 0.0f;
}

// pass 1: per-(edge,head) logit = (q[dst,h] . k_rel[src,h]) * p_rel[h] * SCALE
__global__ void hgt_edge_logits(const int* __restrict__ src, const int* __restrict__ dst,
                                const float* __restrict__ q /* kqv_dst + 64 chunk, ld 192 */,
                                const float* __restrict__ krel /* [Nsrc,64] */,
                                const float* __restrict__ prel /* [4] */,
                                float* __restrict__ elog, int* __restrict__ mx, int E)
{
  int t = blockIdx.x * blockDim.x + threadIdx.x;
  if (t >= E * 4) return;
  const int e = t >> 2, h = t & 3;
  const int s = src[e], d = dst[e];
  const float4* qp = (const float4*)(q + (size_t)d * 192 + h * 16);
  const float4* kp = (const float4*)(krel + (size_t)s * 64 + h * 16);
  float a = 0.f;
  #pragma unroll
  for (int i = 0; i < 4; ++i) {
    const float4 qa = qp[i], ka = kp[i];
    a += qa.x * ka.x + qa.y * ka.y + qa.z * ka.z + qa.w * ka.w;
  }
  a *= prel[h] * 0.25f;                     // SCALE = 1/sqrt(16)
  elog[t] = a;
  atomicMax(mx + d * 4 + h, hgt_enc(a));
}

// pass 2: exp(lg - max) and segment sum
__global__ void hgt_edge_expsum(const int* __restrict__ dst, float* __restrict__ elog,
                                const int* __restrict__ mx, float* __restrict__ sm, int E)
{
  int t = blockIdx.x * blockDim.x + threadIdx.x;
  if (t >= E * 4) return;
  const int e = t >> 2, h = t & 3;
  const int d = dst[e];
  const float ex = expf(elog[t] - hgt_dec(mx[d * 4 + h]));
  elog[t] = ex;
  atomicAdd(sm + d * 4 + h, ex);
}

// pass 3: agg[dst] += att * v_rel[src]
__global__ void hgt_edge_agg(const int* __restrict__ src, const int* __restrict__ dst,
                             const float* __restrict__ elog, const float* __restrict__ sm,
                             const float* __restrict__ vrel, float* __restrict__ agg, int E)
{
  int t = blockIdx.x * blockDim.x + threadIdx.x;
  if (t >= E * 64) return;
  const int e = t >> 6, j = t & 63, h = j >> 4;
  const int s = src[e], d = dst[e];
  const float att = elog[(size_t)e * 4 + h] / fmaxf(sm[d * 4 + h], 1e-16f);
  atomicAdd(agg + (size_t)d * 64 + j, vrel[(size_t)s * 64 + j] * att);
}

// x = sigmoid(skip)*o + (1-sigmoid(skip))*x  (optionally mirror into h-stack)
// vectorized float4; n4 = n/4
__global__ void hgt_blend(const float* __restrict__ o, const float* __restrict__ skip,
                          float* __restrict__ x, float* __restrict__ hout, int n4)
{
  int t = blockIdx.x * blockDim.x + threadIdx.x;
  if (t >= n4) return;
  const float al = 1.0f / (1.0f + expf(-skip[0]));
  const float be = 1.0f - al;
  const float4 ov = ((const float4*)o)[t];
  float4 xv = ((const float4*)x)[t];
  xv.x = al * ov.x + be * xv.x;
  xv.y = al * ov.y + be * xv.y;
  xv.z = al * ov.z + be * xv.z;
  xv.w = al * ov.w + be * xv.w;
  ((float4*)x)[t] = xv;
  if (hout) ((float4*)hout)[t] = xv;
}

// fused jumping-knowledge attention over the layer axis, one node per block (64 thr)
__global__ void hgt_jk(const float* __restrict__ hstack, // [5][NI][64]
                       const float* __restrict__ win,    // [192][64]
                       const float* __restrict__ bin,    // [192]
                       const float* __restrict__ wout,   // [64][64]
                       const float* __restrict__ bout,   // [64]
                       float* __restrict__ hfin)
{
  __shared__ float hb[NLAY][64];
  __shared__ float qkv[NLAY][192];
  __shared__ float ob[NLAY][64];
  __shared__ float attw[HEADS * NLAY][NLAY];
  const int n = blockIdx.x, tid = threadIdx.x;
  for (int l = 0; l < NLAY; ++l) hb[l][tid] = hstack[((size_t)l * NI + n) * 64 + tid];
  __syncthreads();
  for (int idx = tid; idx < NLAY * 192; idx += 64) {
    const int l = idx / 192, c = idx - l * 192;
    float a = bin[c];
    #pragma unroll 8
    for (int d = 0; d < 64; ++d) a += hb[l][d] * win[c * 64 + d];
    qkv[l][c] = a;
  }
  __syncthreads();
  if (tid < HEADS * NLAY) {
    const int h = tid / NLAY, l = tid - h * NLAY;
    float lg[NLAY]; float mxv = -1e30f;
    for (int m = 0; m < NLAY; ++m) {
      float a = 0.f;
      #pragma unroll
      for (int d = 0; d < 16; ++d) a += qkv[l][h * 16 + d] * qkv[m][64 + h * 16 + d];
      a *= 0.25f; lg[m] = a; mxv = fmaxf(mxv, a);
    }
    float s = 0.f;
    for (int m = 0; m < NLAY; ++m) { lg[m] = expf(lg[m] - mxv); s += lg[m]; }
    for (int m = 0; m < NLAY; ++m) attw[tid][m] = lg[m] / s;
  }
  __syncthreads();
  for (int idx = tid; idx < NLAY * 64; idx += 64) {
    const int l = idx >> 6, c = idx & 63, h = c >> 4;
    float a = 0.f;
    for (int m = 0; m < NLAY; ++m) a += attw[h * NLAY + l][m] * qkv[m][128 + c];
    ob[l][c] = a;
  }
  __syncthreads();
  {
    const int f = tid;
    float a = 0.f;
    for (int l = 0; l < NLAY; ++l) {
      float s = 0.f;
      #pragma unroll 8
      for (int d = 0; d < 64; ++d) s += ob[l][d] * wout[f * 64 + d];
      a += s;
    }
    hfin[(size_t)n * 64 + f] = a * 0.2f + bout[f];
  }
}

// per-node: r[n] = h.w_rel ; score[n] = b + h.w_root (edge sums added atomically)
__global__ void hgt_score_init(const float* __restrict__ hfin, const float* __restrict__ wrel,
                               const float* __restrict__ brel, const float* __restrict__ wroot,
                               float* __restrict__ r, float* __restrict__ score, int n)
{
  int i = blockIdx.x * blockDim.x + threadIdx.x;
  if (i >= n) return;
  float a = 0.f, b = 0.f;
  #pragma unroll 8
  for (int f = 0; f < 64; ++f) { const float h = hfin[(size_t)i * 64 + f]; a += h * wrel[f]; b += h * wroot[f]; }
  r[i] = a; score[i] = b + brel[0];
}
__global__ void hgt_score_edges(const int* __restrict__ src, const int* __restrict__ dst,
                                const float* __restrict__ r, float* __restrict__ score, int E)
{
  int e = blockIdx.x * blockDim.x + threadIdx.x;
  if (e < E) atomicAdd(score + dst[e], r[src[e]]);
}

// per-graph iterative top-16 (descending, lowest-index tie-break)
__global__ void hgt_topk(const float* __restrict__ score, int* __restrict__ idxb,
                         float* __restrict__ gate)
{
  __shared__ float sv[NPER];
  __shared__ float rv[256];
  __shared__ int   ri[256];
  const int g = blockIdx.x, tid = threadIdx.x;
  for (int i = tid; i < NPER; i += 256) sv[i] = score[g * NPER + i];
  __syncthreads();
  for (int p = 0; p < POOLK; ++p) {
    float best = -INFINITY; int bi = 0x7fffffff;
    for (int i = tid; i < NPER; i += 256) {
      const float v = sv[i];
      if (v > best || (v == best && i < bi)) { best = v; bi = i; }
    }
    rv[tid] = best; ri[tid] = bi; __syncthreads();
    for (int s = 128; s > 0; s >>= 1) {
      if (tid < s) {
        if (rv[tid + s] > rv[tid] || (rv[tid + s] == rv[tid] && ri[tid + s] < ri[tid])) {
          rv[tid] = rv[tid + s]; ri[tid] = ri[tid + s];
        }
      }
      __syncthreads();
    }
    if (tid == 0) {
      idxb[g * POOLK + p] = g * NPER + ri[0];
      gate[g * POOLK + p] = tanhf(rv[0]);
      sv[ri[0]] = -INFINITY;
    }
    __syncthreads();
  }
}

__global__ void hgt_gather_xs(const float* __restrict__ hfin, const int* __restrict__ idxb,
                              const float* __restrict__ gate, float* __restrict__ xs)
{
  int t = blockIdx.x * blockDim.x + threadIdx.x;
  if (t >= NGR * POOLK * 64) return;
  const int gp = t >> 6, f = t & 63;
  xs[t] = hfin[(size_t)idxb[gp] * 64 + f] * gate[gp];
}

// ---------------- host-side orchestration ----------------
extern "C" void kernel_launch(void* const* d_in, const int* in_sizes, int n_in,
                              void* d_out, int out_size, void* d_ws, size_t ws_size,
                              hipStream_t stream)
{
  (void)in_sizes; (void)n_in; (void)out_size;
  const float* x_instr = (const float*)d_in[IN_X_INSTR];
  const float* x_var   = (const float*)d_in[IN_X_VAR];
  const float* x_base  = (const float*)d_in[IN_X_BASE];
  const float* arel_def  = (const float*)d_in[IN_A_REL_DEF];
  const float* arel_flow = (const float*)d_in[IN_A_REL_FLOW];
  const float* arel_uses = (const float*)d_in[IN_A_REL_USES];
  const float* arel_base = (const float*)d_in[IN_A_REL_BASE];
  const float* mrel_def  = (const float*)d_in[IN_M_REL_DEF];
  const float* mrel_flow = (const float*)d_in[IN_M_REL_FLOW];
  const float* mrel_uses = (const float*)d_in[IN_M_REL_USES];
  const float* mrel_base = (const float*)d_in[IN_M_REL_BASE];
  const float* gnn_rel_b = (const float*)d_in[IN_GNN_REL_B];
  const float* gnn_rel_w = (const float*)d_in[IN_GNN_REL_W];
  const float* gnn_root_w= (const float*)d_in[IN_GNN_ROOT_W];
  const float* jk_in_b  = (const float*)d_in[IN_JK_IN_B];
  const float* jk_in_w  = (const float*)d_in[IN_JK_IN_W];
  const float* jk_out_b = (const float*)d_in[IN_JK_OUT_B];
  const float* jk_out_w = (const float*)d_in[IN_JK_OUT_W];
  const float* kqvb_i   = (const float*)d_in[IN_KQV_B_INSTR];
  const float* kqvb_v   = (const float*)d_in[IN_KQV_B_VAR];
  const float* kqvb_bse = (const float*)d_in[IN_KQV_B_BASE];
  const float* kqvw_i   = (const float*)d_in[IN_KQV_W_INSTR];
  const float* kqvw_v   = (const float*)d_in[IN_KQV_W_VAR];
  const float* kqvw_bse = (const float*)d_in[IN_KQV_W_BASE];
  const float* outb_i   = (const float*)d_in[IN_OUT_B_INSTR];
  const float* outb_v   = (const float*)d_in[IN_OUT_B_VAR];
  const float* outw_i   = (const float*)d_in[IN_OUT_W_INSTR];
  const float* outw_v   = (const float*)d_in[IN_OUT_W_VAR];
  const float* prel_def = (const float*)d_in[IN_P_REL_DEF];
  const float* prel_flow= (const float*)d_in[IN_P_REL_FLOW];
  const float* prel_uses= (const float*)d_in[IN_P_REL_USES];
  const float* prel_base= (const float*)d_in[IN_P_REL_BASE];
  const float* skip_i   = (const float*)d_in[IN_SKIP_INSTR];
  const float* skip_v   = (const float*)d_in[IN_SKIP_VAR];
  const int* ef = (const int*)d_in[IN_EI_FLOW];
  const int* eu = (const int*)d_in[IN_EI_USES];
  const int* ed = (const int*)d_in[IN_EI_DEF];
  const int* eb = (const int*)d_in[IN_EI_BASE];
  const int *ef_s = ef, *ef_d = ef + EFLOW;
  const int *eu_s = eu, *eu_d = eu + EUSES;
  const int *ed_s = ed, *ed_d = ed + EDEF;
  const int *eb_s = eb, *eb_d = eb + EBASE;

  // ---- workspace carve-up (floats) ----
  float* ws = (float*)d_ws;
  size_t o = 0;
  float* cur_xi = ws + o; o += (size_t)NI * 64;
  float* cur_xv = ws + o; o += (size_t)NV * 64;
  float* kqv_i  = ws + o; o += (size_t)NI * 192;
  float* kqv_v  = ws + o; o += (size_t)NV * 192;
  float* kqv_b  = ws + o; o += (size_t)NBASE * 192;
  float* kr_flow = ws + o; o += (size_t)NI * 64;
  float* vr_flow = ws + o; o += (size_t)NI * 64;
  float* kr_uses = ws + o; o += (size_t)NI * 64;
  float* vr_uses = ws + o; o += (size_t)NI * 64;
  float* kr_def  = ws + o; o += (size_t)NV * 64;
  float* vr_def  = ws + o; o += (size_t)NV * 64;
  float* kr_base = ws + o; o += (size_t)NBASE * 64;
  float* vr_base = ws + o; o += (size_t)NBASE * 64;
  float* elog = ws + o; o += (size_t)(EFLOW + EDEF + EBASE) * 4;
  int*   mx   = (int*)(ws + o); o += (size_t)NI * 4;
  float* sm   = ws + o; o += (size_t)NI * 4;
  float* agg  = ws + o; o += (size_t)NI * 64;
  float* obuf = ws + o; o += (size_t)NI * 64;
  float* hstack = ws + o; o += (size_t)NLAY * NI * 64;
  float* hfin  = ws + o; o += (size_t)NI * 64;
  float* rbuf  = ws + o; o += (size_t)NI;
  float* scoreb= ws + o; o += (size_t)NI;
  int*   idxb  = (int*)(ws + o); o += (size_t)NGR * POOLK;
  float* gateb = ws + o; o += (size_t)NGR * POOLK;
  float* xsb   = ws + o; o += (size_t)NGR * NPER;
  float* mt0   = ws + o; o += (size_t)64 * 1024;
  float* mt1   = ws + o; o += (size_t)64 * 512;
  float* wbd   = ws + o; o += (size_t)64 * 64;     // block-diagonal rel weight scratch
  if (ws_size < o * sizeof(float)) return;   // workspace too small: bail deterministically

  auto gemm = [&](const float* A, int lda, const float* W, int ldw, const float* bias,
                  float* C, int ldc, int M, int K, int N) {
    const int nT = ((M + 15) / 16) * ((N + 15) / 16);
    const int wpb = 8;                       // 256 threads = 8 wave32s
    const int grid = (nT + wpb - 1) / wpb;
    const bool fast = (M % 16 == 0) && (N % 16 == 0) && (K % 32 == 0);
    if (fast) hipLaunchKernelGGL((hgt_gemm_wmma<false>), dim3(grid), dim3(256), 0, stream, A, lda, W, ldw, bias, C, ldc, M, K, N);
    else      hipLaunchKernelGGL((hgt_gemm_wmma<true>),  dim3(grid), dim3(256), 0, stream, A, lda, W, ldw, bias, C, ldc, M, K, N);
  };
  auto fillf = [&](float* p, float v, int n) { hgt_fill_f32<<<(n + 255) / 256, 256, 0, stream>>>(p, v, n); };
  auto filli = [&](int* p, int v, int n)     { hgt_fill_i32<<<(n + 255) / 256, 256, 0, stream>>>(p, v, n); };
  auto geluip = [&](float* p, int n) {       // n must be a multiple of 4
    const int n4 = n >> 2;
    hgt_gelu_inplace<<<(n4 + 255) / 256, 256, 0, stream>>>(p, n4);
  };
  // per-head 16x16 rel transform == one K=64 GEMM against blockdiag weight
  auto reltrans = [&](const float* kqv_src, int chunk /*0=k,128=v*/, const float* relW,
                      float* out, int Mnodes) {
    hgt_blockdiag<<<16, 256, 0, stream>>>(relW, wbd);
    gemm(kqv_src + chunk, 192, wbd, 64, nullptr, out, 64, Mnodes, 64, 64);
  };

  hipMemcpyAsync(cur_xi, x_instr, (size_t)NI * 64 * sizeof(float), hipMemcpyDeviceToDevice, stream);
  hipMemcpyAsync(cur_xv, x_var,   (size_t)NV * 64 * sizeof(float), hipMemcpyDeviceToDevice, stream);

  for (int l = 0; l < NLAY; ++l) {
    // fused K|Q|V projection (k cols 0..63, q 64..127, v 128..191)
    gemm(cur_xi, 64, kqvw_i + (size_t)l * 64 * 192, 192, kqvb_i + (size_t)l * 192, kqv_i, 192, NI, 64, 192);
    gemm(cur_xv, 64, kqvw_v + (size_t)l * 64 * 192, 192, kqvb_v + (size_t)l * 192, kqv_v, 192, NV, 64, 192);
    if (l == 0)
      gemm(x_base, 64, kqvw_bse, 192, kqvb_bse, kqv_b, 192, NBASE, 64, 192);

    // hoisted per-edge rel transforms: per-source-node block-diagonal WMMA GEMMs
    const size_t lw = (size_t)l * HEADS * 256;
    reltrans(kqv_i, 0,   arel_flow + lw, kr_flow, NI);
    reltrans(kqv_i, 128, mrel_flow + lw, vr_flow, NI);
    reltrans(kqv_i, 0,   arel_uses + lw, kr_uses, NI);
    reltrans(kqv_i, 128, mrel_uses + lw, vr_uses, NI);
    reltrans(kqv_v, 0,   arel_def  + lw, kr_def,  NV);
    reltrans(kqv_v, 128, mrel_def  + lw, vr_def,  NV);
    if (l == 0) {
      reltrans(kqv_b, 0,   arel_base, kr_base, NBASE);
      reltrans(kqv_b, 128, mrel_base, vr_base, NBASE);
    }

    // ===== dst = instr (flow + def [+ base at l==0]), shared segment softmax =====
    filli(mx, (int)0x80000000, NI * 4);
    fillf(sm, 0.f, NI * 4);
    fillf(agg, 0.f, NI * 64);
    float* el_f = elog;
    float* el_d = elog + (size_t)EFLOW * 4;
    float* el_b = elog + (size_t)(EFLOW + EDEF) * 4;
    hgt_edge_logits<<<(EFLOW * 4 + 255) / 256, 256, 0, stream>>>(ef_s, ef_d, kqv_i + 64, kr_flow, prel_flow + l * 4, el_f, mx, EFLOW);
    hgt_edge_logits<<<(EDEF  * 4 + 255) / 256, 256, 0, stream>>>(ed_s, ed_d, kqv_i + 64, kr_def,  prel_def  + l * 4, el_d, mx, EDEF);
    if (l == 0)
      hgt_edge_logits<<<(EBASE * 4 + 255) / 256, 256, 0, stream>>>(eb_s, eb_d, kqv_i + 64, kr_base, prel_base, el_b, mx, EBASE);
    hgt_edge_expsum<<<(EFLOW * 4 + 255) / 256, 256, 0, stream>>>(ef_d, el_f, mx, sm, EFLOW);
    hgt_edge_expsum<<<(EDEF  * 4 + 255) / 256, 256, 0, stream>>>(ed_d, el_d, mx, sm, EDEF);
    if (l == 0)
      hgt_edge_expsum<<<(EBASE * 4 + 255) / 256, 256, 0, stream>>>(eb_d, el_b, mx, sm, EBASE);
    hgt_edge_agg<<<(EFLOW * 64 + 255) / 256, 256, 0, stream>>>(ef_s, ef_d, el_f, sm, vr_flow, agg, EFLOW);
    hgt_edge_agg<<<(EDEF  * 64 + 255) / 256, 256, 0, stream>>>(ed_s, ed_d, el_d, sm, vr_def,  agg, EDEF);
    if (l == 0)
      hgt_edge_agg<<<(EBASE * 64 + 255) / 256, 256, 0, stream>>>(eb_s, eb_d, el_b, sm, vr_base, agg, EBASE);
    // o = gelu(agg) @ out_w + out_b  (GELU hoisted to one vector pass; agg is scratch)
    geluip(agg, NI * 64);
    gemm(agg, 64, outw_i + (size_t)l * 64 * 64, 64, outb_i + (size_t)l * 64, obuf, 64, NI, 64, 64);
    hgt_blend<<<(NI * 16 + 255) / 256, 256, 0, stream>>>(obuf, skip_i + l, cur_xi, hstack + (size_t)l * NI * 64, NI * 16);

    // ===== dst = var (uses) =====
    filli(mx, (int)0x80000000, NV * 4);
    fillf(sm, 0.f, NV * 4);
    fillf(agg, 0.f, NV * 64);
    hgt_edge_logits<<<(EUSES * 4 + 255) / 256, 256, 0, stream>>>(eu_s, eu_d, kqv_v + 64, kr_uses, prel_uses + l * 4, elog, mx, EUSES);
    hgt_edge_expsum<<<(EUSES * 4 + 255) / 256, 256, 0, stream>>>(eu_d, elog, mx, sm, EUSES);
    hgt_edge_agg<<<(EUSES * 64 + 255) / 256, 256, 0, stream>>>(eu_s, eu_d, elog, sm, vr_uses, agg, EUSES);
    geluip(agg, NV * 64);
    gemm(agg, 64, outw_v + (size_t)l * 64 * 64, 64, outb_v + (size_t)l * 64, obuf, 64, NV, 64, 64);
    hgt_blend<<<(NV * 16 + 255) / 256, 256, 0, stream>>>(obuf, skip_v + l, cur_xv, nullptr, NV * 16);
  }

  // jumping-knowledge attention over layers (fused, LDS-resident per node)
  hgt_jk<<<NI, 64, 0, stream>>>(hstack, jk_in_w, jk_in_b, jk_out_w, jk_out_b, hfin);

  // SAGPool scoring via linearity: segsum(h[src])·w == segsum(h[src]·w)
  hgt_score_init<<<(NI + 255) / 256, 256, 0, stream>>>(hfin, gnn_rel_w, gnn_rel_b, gnn_root_w, rbuf, scoreb, NI);
  hgt_score_edges<<<(EFLOW + 255) / 256, 256, 0, stream>>>(ef_s, ef_d, rbuf, scoreb, EFLOW);
  hgt_topk<<<NGR, 256, 0, stream>>>(scoreb, idxb, gateb);
  hgt_gather_xs<<<(NGR * POOLK * 64 + 255) / 256, 256, 0, stream>>>(hfin, idxb, gateb, xsb);

  // MLP head: 1024->512->...->2->1; GELU applied in-place on scratch activations
  const int mdims[11] = {1024, 512, 256, 128, 64, 32, 16, 8, 4, 2, 1};
  const float* cur = xsb;
  for (int i = 0; i < 10; ++i) {
    const float* W = (const float*)d_in[IN_MLP_W0 + i];
    const float* bb = (const float*)d_in[IN_MLP_B0 + i];
    float* out = (i == 9) ? (float*)d_out : ((i & 1) ? mt1 : mt0);
    if (i > 0) geluip((float*)cur, NGR * mdims[i]);   // cur is always scratch for i>0
    gemm(cur, mdims[i], W, mdims[i + 1], bb, out, mdims[i + 1], NGR, mdims[i], mdims[i + 1]);
    cur = out;
  }
}
